// TransformerDecoderBlock_10307921510770
// MI455X (gfx1250) — matrix operand order repaired and checked by reading, not compiled, not run on hardware
//
#include <hip/hip_runtime.h>
#include <hip/hip_bf16.h>

// ---------------------------------------------------------------------------
// Types for CDNA5 WMMA (wave32): V_WMMA_F32_16X16X32_BF16
// ---------------------------------------------------------------------------
typedef __bf16 v16bf __attribute__((ext_vector_type(16)));
typedef __bf16 v8bf  __attribute__((ext_vector_type(8)));
typedef float  v8f   __attribute__((ext_vector_type(8)));

__device__ __forceinline__ v8f wmma_bf16(v16bf a, v16bf b, v8f c) {
#if defined(__HIP_DEVICE_COMPILE__)
    // 8 args: (neg_a, A, neg_b, B, c_mod, C, reuse_a, reuse_b)
    return __builtin_amdgcn_wmma_f32_16x16x32_bf16(false, a, false, b,
                                                   (short)0, c, false, false);
#else
    return c;
#endif
}

// CDNA5 async memory->LDS staging (ASYNCcnt path, §10 GLOBAL_LOAD_ASYNC_TO_LDS).
// GV mode: VDST = per-lane LDS byte offset, VADDR = 64-bit global address.
__device__ __forceinline__ void async_copy_b128(void* lds, const void* gptr) {
#if defined(__HIP_DEVICE_COMPILE__)
    unsigned loff = (unsigned)(size_t)lds;     // flat LDS ptr low 32b = LDS offset
    asm volatile("global_load_async_to_lds_b128 %0, %1, off"
                 :: "v"(loff), "v"(gptr) : "memory");
#else
    (void)lds; (void)gptr;
#endif
}
__device__ __forceinline__ void async_wait0() {
#if defined(__HIP_DEVICE_COMPILE__)
    asm volatile("s_wait_asynccnt 0x0" ::: "memory");
#endif
}

// A-matrix 16x32 bf16 fragment from row-major [16][>=64] LDS tile.
// Lane l: row = l&15; K elements j<8 -> 8*(l>>4)+j ; j>=8 -> 16+8*(l>>4)+(j-8)
__device__ __forceinline__ v16bf load_a_frag(const __bf16* base, int ld) {
    int lane = threadIdx.x & 31;
    const __bf16* p = base + (lane & 15) * ld + ((lane >> 4) << 3);
    v8bf lo = *(const v8bf*)(p);
    v8bf hi = *(const v8bf*)(p + 16);
    return __builtin_shufflevector(lo, hi, 0,1,2,3,4,5,6,7,8,9,10,11,12,13,14,15);
}

// B-matrix 32x16 bf16 fragment from [N][K] LDS tile (K contiguous).
// Lane l: col N = l&15 ; K elements j<8 -> 8*(l>>4)+j ; j>=8 -> 16+8*(l>>4)+(j-8)
__device__ __forceinline__ v16bf load_b_frag(const __bf16* base, int ld) {
    int lane = threadIdx.x & 31;
    const __bf16* p = base + (lane & 15) * ld + ((lane >> 4) << 3);
    v8bf lo = *(const v8bf*)(p);
    v8bf hi = *(const v8bf*)(p + 16);
    return __builtin_shufflevector(lo, hi, 0,1,2,3,4,5,6,7,8,9,10,11,12,13,14,15);
}

// ---------------------------------------------------------------------------
// Problem constants (fixed by the reference)
// ---------------------------------------------------------------------------
#define BB 2
#define SS 2048
#define DD 1024
#define HH 16
#define HD 64
#define ROWS (BB*SS)      // 4096
#define LDP 80            // padded LDS row stride (160B: 16B aligned, bank-spread)

// ---------------------------------------------------------------------------
// Weight prep: fp32 -> bf16, transposed to [N][K]
// ---------------------------------------------------------------------------
__global__ void prep_qkvt(const float* __restrict__ Wq, const float* __restrict__ Wk,
                          const float* __restrict__ Wv, __bf16* __restrict__ Wt) {
    int idx = blockIdx.x * 256 + threadIdx.x;      // over 3072*1024
    int n = idx >> 10, d = idx & 1023;
    const float* W = (n < 1024) ? Wq : ((n < 2048) ? Wk : Wv);
    int nh = n & 1023, h = nh >> 6, kk = nh & 63;  // W[h][d][kk], shape [H,D,hd]
    Wt[idx] = (__bf16)W[(h * DD + d) * HD + kk];
}

__global__ void transpose_bf16(const float* __restrict__ W, __bf16* __restrict__ Wt,
                               int K, int N) {
    int idx = blockIdx.x * 256 + threadIdx.x;      // over N*K
    if (idx >= N * K) return;
    int n = idx / K, kk = idx - n * K;
    Wt[idx] = (__bf16)W[kk * N + n];               // Wt[n][k] = W[k][n]
}

// ---------------------------------------------------------------------------
// LayerNorm (fp32 in) -> bf16 out, one block per row, D = 1024
// ---------------------------------------------------------------------------
__global__ __launch_bounds__(256) void ln_to_bf16(const float* __restrict__ x,
                                                  const float* __restrict__ g,
                                                  const float* __restrict__ b,
                                                  __bf16* __restrict__ out) {
    int row = blockIdx.x;
    const float* xr = x + (long)row * DD;
    float v[4], s = 0.f, ss = 0.f;
    for (int i = 0; i < 4; ++i) {
        v[i] = xr[threadIdx.x + i * 256];
        s += v[i]; ss += v[i] * v[i];
    }
    __shared__ float r0[256], r1[256];
    r0[threadIdx.x] = s; r1[threadIdx.x] = ss;
    __syncthreads();
    for (int off = 128; off > 0; off >>= 1) {
        if (threadIdx.x < off) {
            r0[threadIdx.x] += r0[threadIdx.x + off];
            r1[threadIdx.x] += r1[threadIdx.x + off];
        }
        __syncthreads();
    }
    float mu  = r0[0] * (1.f / DD);
    float var = r1[0] * (1.f / DD) - mu * mu;
    float rs  = rsqrtf(var + 1e-5f);
    for (int i = 0; i < 4; ++i) {
        int c = threadIdx.x + i * 256;
        out[(long)row * DD + c] = (__bf16)((v[i] - mu) * rs * g[c] + b[c]);
    }
}

// ---------------------------------------------------------------------------
// Generic GEMM: C[M,N] = A_bf16[M,K] * Bt_bf16[N,K]^T
// Block tile 128x128, 8 waves (4m x 2n) of 32x64, K-step 64 -> 16 WMMA/wave/step.
// Staging via global_load_async_to_lds_b128 (ASYNCcnt).
// EPI 0: bf16 out ; EPI 1: fp32 out = resid + acc
// ---------------------------------------------------------------------------
template <int EPI>
__global__ __launch_bounds__(256) void gemm_bt(const __bf16* __restrict__ A,
                                               const __bf16* __restrict__ Bt,
                                               float* __restrict__ Cf,
                                               __bf16* __restrict__ Cb,
                                               const float* __restrict__ resid,
                                               int M, int N, int K) {
    __shared__ __bf16 As[128][LDP];
    __shared__ __bf16 Bs[128][LDP];
    int tid = threadIdx.x, lane = tid & 31, w = tid >> 5;
    int wm = w >> 1, wn = w & 1;
    int m0 = blockIdx.y * 128, n0 = blockIdx.x * 128;
    v8f acc[2][4] = {};
    for (int k0 = 0; k0 < K; k0 += 64) {
        __syncthreads();                                   // LDS reuse guard
#pragma unroll
        for (int i = 0; i < 4; ++i) {                      // 1024 chunks each
            int c = tid + i * 256;
            int r = c >> 3, col = (c & 7) * 8;
            async_copy_b128(&As[r][col], &A[(long)(m0 + r) * K + k0 + col]);
            async_copy_b128(&Bs[r][col], &Bt[(long)(n0 + r) * K + k0 + col]);
        }
        async_wait0();
        __syncthreads();
#pragma unroll
        for (int s = 0; s < 2; ++s) {
            v16bf af[2], bfr[4];
#pragma unroll
            for (int i = 0; i < 2; ++i)
                af[i] = load_a_frag(&As[wm * 32 + i * 16][s * 32], LDP);
#pragma unroll
            for (int j = 0; j < 4; ++j)
                bfr[j] = load_b_frag(&Bs[wn * 64 + j * 16][s * 32], LDP);
#pragma unroll
            for (int i = 0; i < 2; ++i)
#pragma unroll
                for (int j = 0; j < 4; ++j)
                    acc[i][j] = wmma_bf16(af[i], bfr[j], acc[i][j]);
        }
    }
    int half = lane >> 4, coln = lane & 15;
#pragma unroll
    for (int i = 0; i < 2; ++i)
#pragma unroll
        for (int j = 0; j < 4; ++j)
#pragma unroll
            for (int r = 0; r < 8; ++r) {
                int row = m0 + wm * 32 + i * 16 + half * 8 + r;
                int col = n0 + wn * 64 + j * 16 + coln;
                long idx = (long)row * N + col;
                float v = acc[i][j][r];
                if (EPI == 0) Cb[idx] = (__bf16)v;
                else          Cf[idx] = resid[idx] + v;
            }
}

// ---------------------------------------------------------------------------
// RoPE + split: qkv bf16 [4096][3072] -> q[BH][S][hd] (pre-scaled by hd^-0.5),
// k[BH][S][hd], vT[BH][hd][S]. theta = 10000^(-i/16), faithful to reference.
// ---------------------------------------------------------------------------
__global__ __launch_bounds__(256) void rope_split(const __bf16* __restrict__ qkv,
                                                  __bf16* __restrict__ q,
                                                  __bf16* __restrict__ k,
                                                  __bf16* __restrict__ vt) {
    int row = blockIdx.x;                  // b*S + s
    int b = row >> 11, s = row & 2047;
    const __bf16* src = qkv + (long)row * 3072;
    for (int p = threadIdx.x; p < 512; p += 256) {   // 512 (head,pair) combos
        int h = p >> 5, i = p & 31;
        float theta = __powf(10000.f, -(float)i * (1.f / 16.f));
        float ang = (float)s * theta;
        float sn, cs; __sincosf(ang, &sn, &cs);
        int base = ((b * HH + h) * SS + s) * HD + 2 * i;
        float x0 = (float)src[h * HD + 2 * i], x1 = (float)src[h * HD + 2 * i + 1];
        q[base]     = (__bf16)(0.125f * (x0 * cs - x1 * sn));   // fold hd^-0.5
        q[base + 1] = (__bf16)(0.125f * (x1 * cs + x0 * sn));
        float y0 = (float)src[DD + h * HD + 2 * i], y1 = (float)src[DD + h * HD + 2 * i + 1];
        k[base]     = (__bf16)(y0 * cs - y1 * sn);
        k[base + 1] = (__bf16)(y1 * cs + y0 * sn);
    }
    for (int d = threadIdx.x; d < DD; d += 256) {    // V transposed: [hd][S]
        int h = d >> 6, dd = d & 63;
        vt[((long)(b * HH + h) * HD + dd) * SS + s] = src[2048 + d];
    }
}

// ---------------------------------------------------------------------------
// Flash attention, causal. One block = (b,h) x 64-query tile. 4 waves,
// each wave owns 16 query rows. Online softmax stats per-lane: C-tile layout
// => lane group (l>>4) holds rows r + 8*(l>>4); shfl_xor(width 16) row-reduce.
// K/V tiles staged via async-to-LDS.
// ---------------------------------------------------------------------------
__global__ __launch_bounds__(128) void flash_attn(const __bf16* __restrict__ q,
                                                  const __bf16* __restrict__ k,
                                                  const __bf16* __restrict__ vt,
                                                  __bf16* __restrict__ attn_out) {
    int bh = blockIdx.x;                // 0..31  (b*H + h)
    int qt = blockIdx.y;                // 0..31  query tile
    int b = bh >> 4, h = bh & 15;
    int q0 = qt * 64;
    __shared__ __bf16 Qs[64][LDP];
    __shared__ __bf16 Ks[64][LDP];
    __shared__ __bf16 Vs[64][LDP];       // Vs[d][key]
    __shared__ __bf16 Ps[4][16][LDP];    // per-wave P tile
    int tid = threadIdx.x, lane = tid & 31, w = tid >> 5;
    int half = lane >> 4, coln = lane & 15;

    const __bf16* qbase = q + ((long)bh * SS + q0) * HD;
#pragma unroll
    for (int i = 0; i < 4; ++i) {                    // 512 chunks of 16B
        int c = tid + i * 128;
        int r = c >> 3, col = (c & 7) * 8;
        async_copy_b128(&Qs[r][col], &qbase[r * HD + col]);
    }
    async_wait0();
    __syncthreads();
    v16bf qf0 = load_a_frag(&Qs[w * 16][0], LDP);
    v16bf qf1 = load_a_frag(&Qs[w * 16][32], LDP);

    v8f oacc[4] = {};
    float mrow[8], lrow[8];
#pragma unroll
    for (int r = 0; r < 8; ++r) { mrow[r] = -3.0e38f; lrow[r] = 0.f; }

    for (int kb = 0; kb <= qt; ++kb) {
        __syncthreads();                                   // protect Ks/Vs reuse
        const __bf16* kbase = k + ((long)bh * SS + kb * 64) * HD;
        const __bf16* vbase = vt + (long)bh * HD * SS + kb * 64;
#pragma unroll
        for (int i = 0; i < 4; ++i) {
            int c = tid + i * 128;
            int r = c >> 3, col = (c & 7) * 8;
            async_copy_b128(&Ks[r][col], &kbase[r * HD + col]);
            async_copy_b128(&Vs[r][col], &vbase[(long)r * SS + col]);
        }
        async_wait0();
        __syncthreads();

        // S = Q * K^T   (Q pre-scaled by hd^-0.5)
        v8f sacc[4];
#pragma unroll
        for (int nj = 0; nj < 4; ++nj) {
            v8f z = {};
            z = wmma_bf16(qf0, load_b_frag(&Ks[nj * 16][0], LDP), z);
            z = wmma_bf16(qf1, load_b_frag(&Ks[nj * 16][32], LDP), z);
            sacc[nj] = z;
        }

        // mask + online softmax + write P (bf16) to LDS
#pragma unroll
        for (int r = 0; r < 8; ++r) {
            int qglob = q0 + w * 16 + half * 8 + r;
            float pv[4], mloc = -3.0e38f;
#pragma unroll
            for (int nj = 0; nj < 4; ++nj) {
                int kglob = kb * 64 + nj * 16 + coln;
                float sv = (kglob <= qglob) ? sacc[nj][r] : -3.0e38f;
                pv[nj] = sv;
                mloc = fmaxf(mloc, sv);
            }
            for (int off = 8; off; off >>= 1)
                mloc = fmaxf(mloc, __shfl_xor(mloc, off, 16));
            float mnew = fmaxf(mrow[r], mloc);
            float corr = __expf(mrow[r] - mnew);
            float rsum = 0.f;
#pragma unroll
            for (int nj = 0; nj < 4; ++nj) {
                float p = __expf(pv[nj] - mnew);
                rsum += p;
                Ps[w][half * 8 + r][nj * 16 + coln] = (__bf16)p;
            }
            for (int off = 8; off; off >>= 1)
                rsum += __shfl_xor(rsum, off, 16);
            lrow[r] = lrow[r] * corr + rsum;
            mrow[r] = mnew;
#pragma unroll
            for (int tj = 0; tj < 4; ++tj) oacc[tj][r] *= corr;
        }

        // O += P * V   (Vs is [hd][key] => B-frag [N=hd][K=key])
        v16bf pf0 = load_a_frag(&Ps[w][0][0], LDP);
        v16bf pf1 = load_a_frag(&Ps[w][0][32], LDP);
#pragma unroll
        for (int tj = 0; tj < 4; ++tj) {
            oacc[tj] = wmma_bf16(pf0, load_b_frag(&Vs[tj * 16][0], LDP), oacc[tj]);
            oacc[tj] = wmma_bf16(pf1, load_b_frag(&Vs[tj * 16][32], LDP), oacc[tj]);
        }
    }

    // normalize and write [B,S,H*hd] bf16 (head-concat layout for O-proj GEMM)
#pragma unroll
    for (int r = 0; r < 8; ++r) {
        float inv = 1.f / lrow[r];
        int srow = q0 + w * 16 + half * 8 + r;
        __bf16* orow = attn_out + ((long)(b * SS + srow)) * DD + h * HD;
#pragma unroll
        for (int tj = 0; tj < 4; ++tj)
            orow[tj * 16 + coln] = (__bf16)(oacc[tj][r] * inv);
    }
}

// ---------------------------------------------------------------------------
// SwiGLU: gate[m][j] = silu(u12[m][j]) * u12[m][2048+j]
// ---------------------------------------------------------------------------
__global__ __launch_bounds__(256) void swiglu(const __bf16* __restrict__ u12,
                                              __bf16* __restrict__ gate) {
    int idx = blockIdx.x * 256 + threadIdx.x;      // over 4096*2048
    int m = idx >> 11, j = idx & 2047;
    float u  = (float)u12[(long)m * 4096 + j];
    float u2 = (float)u12[(long)m * 4096 + 2048 + j];
    float sig = 1.f / (1.f + __expf(-u));
    gate[idx] = (__bf16)(u * sig * u2);
}

// ---------------------------------------------------------------------------
// Orchestration
// ---------------------------------------------------------------------------
extern "C" void kernel_launch(void* const* d_in, const int* in_sizes, int n_in,
                              void* d_out, int out_size, void* d_ws, size_t ws_size,
                              hipStream_t stream) {
    const float* x     = (const float*)d_in[0];
    const float* ln1_g = (const float*)d_in[1];
    const float* ln1_b = (const float*)d_in[2];
    const float* Wq    = (const float*)d_in[3];
    const float* Wk    = (const float*)d_in[4];
    const float* Wv    = (const float*)d_in[5];
    const float* Wo    = (const float*)d_in[6];
    const float* ln2_g = (const float*)d_in[7];
    const float* ln2_b = (const float*)d_in[8];
    const float* W1    = (const float*)d_in[9];
    const float* W2    = (const float*)d_in[10];
    const float* W3    = (const float*)d_in[11];
    float* out = (float*)d_out;

    char* ws = (char*)d_ws;
    size_t off = 0;
    auto alloc = [&](size_t bytes) -> void* {
        void* p = ws + off;
        off = (off + bytes + 255) & ~(size_t)255;
        return p;
    };
    __bf16* wqkvt = (__bf16*)alloc((size_t)3072 * 1024 * 2);   // [3072][1024]
    __bf16* wot   = (__bf16*)alloc((size_t)1024 * 1024 * 2);   // [1024][1024]
    __bf16* w12t  = (__bf16*)alloc((size_t)4096 * 1024 * 2);   // [4096][1024]
    __bf16* w3t   = (__bf16*)alloc((size_t)1024 * 2048 * 2);   // [1024][2048]
    __bf16* h1    = (__bf16*)alloc((size_t)ROWS * DD * 2);
    __bf16* qkv   = (__bf16*)alloc((size_t)ROWS * 3072 * 2);
    __bf16* qb    = (__bf16*)alloc((size_t)ROWS * DD * 2);
    __bf16* kb    = (__bf16*)alloc((size_t)ROWS * DD * 2);
    __bf16* vtb   = (__bf16*)alloc((size_t)ROWS * DD * 2);
    __bf16* attn  = (__bf16*)alloc((size_t)ROWS * DD * 2);
    float*  x1    = (float*)alloc((size_t)ROWS * DD * 4);
    __bf16* h2    = (__bf16*)alloc((size_t)ROWS * DD * 2);
    __bf16* u12   = (__bf16*)alloc((size_t)ROWS * 4096 * 2);
    __bf16* gate  = (__bf16*)alloc((size_t)ROWS * 2048 * 2);
    (void)ws_size; (void)in_sizes; (void)n_in; (void)out_size;

    // --- weight prep (bf16, B-transposed) ---
    prep_qkvt<<<(3072 * 1024) / 256, 256, 0, stream>>>(Wq, Wk, Wv, wqkvt);
    transpose_bf16<<<(1024 * 1024) / 256, 256, 0, stream>>>(Wo, wot, 1024, 1024);
    transpose_bf16<<<(2048 * 1024) / 256, 256, 0, stream>>>(W1, w12t, 1024, 2048);
    transpose_bf16<<<(2048 * 1024) / 256, 256, 0, stream>>>(W2, w12t + (size_t)2048 * 1024, 1024, 2048);
    transpose_bf16<<<(2048 * 1024) / 256, 256, 0, stream>>>(W3, w3t, 2048, 1024);

    // --- attention sublayer ---
    ln_to_bf16<<<ROWS, 256, 0, stream>>>(x, ln1_g, ln1_b, h1);
    gemm_bt<0><<<dim3(3072 / 128, ROWS / 128), 256, 0, stream>>>(
        h1, wqkvt, nullptr, qkv, nullptr, ROWS, 3072, 1024);
    rope_split<<<ROWS, 256, 0, stream>>>(qkv, qb, kb, vtb);
    flash_attn<<<dim3(BB * HH, SS / 64), 128, 0, stream>>>(qb, kb, vtb, attn);
    gemm_bt<1><<<dim3(1024 / 128, ROWS / 128), 256, 0, stream>>>(
        attn, wot, x1, nullptr, x, ROWS, 1024, 1024);

    // --- SwiGLU FFN sublayer ---
    ln_to_bf16<<<ROWS, 256, 0, stream>>>(x1, ln2_g, ln2_b, h2);
    gemm_bt<0><<<dim3(4096 / 128, ROWS / 128), 256, 0, stream>>>(
        h2, w12t, nullptr, u12, nullptr, ROWS, 4096, 1024);
    swiglu<<<(ROWS * 2048) / 256, 256, 0, stream>>>(u12, gate);
    gemm_bt<1><<<dim3(1024 / 128, ROWS / 128), 256, 0, stream>>>(
        gate, w3t, out, nullptr, x1, ROWS, 1024, 2048);
}
